// NestLinearQuantizer_30459908063606
// MI455X (gfx1250) — 8-bit matrix operand order repaired and checked by reading, compiled, not run on hardware
//
#include <hip/hip_runtime.h>
#include <hip/hip_bf16.h>

#define TOK  256
#define KIN  8192
#define NOUT 8192

#define AS1 __attribute__((address_space(1)))
#define AS3 __attribute__((address_space(3)))

typedef int   v4i __attribute__((ext_vector_type(4)));
typedef int   v8i __attribute__((ext_vector_type(8)));
typedef float v4f __attribute__((ext_vector_type(4)));

__device__ __forceinline__ int clampi(int v, int lo, int hi) {
    return v < lo ? lo : (v > hi ? hi : v);
}

__device__ __forceinline__ int pack_q4(int ph0, int ph1, v4i ql) {
    int q0 = ((ph0 >> 4) & 0xF) * 16 + (ql.x & 31) - 144;
    int q1 = (ph0 & 0xF)        * 16 + (ql.y & 31) - 144;
    int q2 = ((ph1 >> 4) & 0xF) * 16 + (ql.z & 31) - 144;
    int q3 = (ph1 & 0xF)        * 16 + (ql.w & 31) - 144;
    q0 = q0 < -128 ? -128 : q0;
    q1 = q1 < -128 ? -128 : q1;
    q2 = q2 < -128 ? -128 : q2;
    q3 = q3 < -128 ? -128 : q3;
    return (q0 & 0xFF) | ((q1 & 0xFF) << 8) | ((q2 & 0xFF) << 16) | ((q3 & 0xFF) << 24);
}

__global__ __launch_bounds__(256)
void quant_x_kernel(const float* __restrict__ x, const float* __restrict__ act_scale,
                    int* __restrict__ a8, int* __restrict__ ssum) {
    __shared__ int red[256];
    const int tok = blockIdx.x;
    const int tid = threadIdx.x;
    const float inv = 1.0f / act_scale[0];
    const v4f* __restrict__ xr = (const v4f*)(x + (size_t)tok * KIN) + (size_t)tid * 8;

    int acc = 0;
    int dw[8];
#pragma unroll
    for (int v = 0; v < 8; ++v) {
        v4f f = xr[v];
        int q0 = clampi((int)__builtin_rintf(f.x * inv), -128, 127);
        int q1 = clampi((int)__builtin_rintf(f.y * inv), -128, 127);
        int q2 = clampi((int)__builtin_rintf(f.z * inv), -128, 127);
        int q3 = clampi((int)__builtin_rintf(f.w * inv), -128, 127);
        acc += q0 + q1 + q2 + q3;
        dw[v] = (q0 & 0xFF) | ((q1 & 0xFF) << 8) | ((q2 & 0xFF) << 16) | ((q3 & 0xFF) << 24);
    }

    const int s    = tid >> 1;
    const int c    = tid & 1;
    const int tile = tok >> 4;
    const int ml   = tok & 15;
    int* base = a8 + (size_t)(s * 16 + tile) * 256;
#pragma unroll
    for (int h = 0; h < 2; ++h) {
        v4i st = {dw[4 * h], dw[4 * h + 1], dw[4 * h + 2], dw[4 * h + 3]};
        *(v4i*)(base + (ml + 16 * h) * 8 + 4 * c) = st;
    }

    red[tid] = acc;
    __syncthreads();
    for (int r = 128; r > 0; r >>= 1) {
        if (tid < r) red[tid] += red[tid + r];
        __syncthreads();
    }
    if (tid == 0) ssum[tok] = red[0];
}

__device__ __forceinline__ void wait_async_le8() {
    asm volatile("s_wait_asynccnt 0x8" ::: "memory");
}
__device__ __forceinline__ void wait_async_0() {
    asm volatile("s_wait_asynccnt 0x0" ::: "memory");
}

__device__ __forceinline__ void issue_slab(const int* a8, int slab, int tid, int* lbuf) {
#pragma unroll
    for (int r = 0; r < 8; ++r) {
        const int* g = a8 + (size_t)slab * 4096 + tid * 4 + r * 512;
        int* l = lbuf + tid * 4 + r * 512;
        __builtin_amdgcn_global_load_async_to_lds_b128(
            (AS1 v4i*)(uintptr_t)g,
            (AS3 v4i*)(unsigned)(uintptr_t)l,
            0, 0);
    }
}

__device__ __forceinline__ void load_w(const v4i* __restrict__ ph_row,
                                       const v4i* __restrict__ pl_row,
                                       int k0, int g, v4i (&lo)[8], v4i (&hh)[4]) {
    const int plbase = k0 >> 2;
    const int phbase = k0 >> 3;
#pragma unroll
    for (int v = 0; v < 8; ++v)
        lo[v] = __builtin_nontemporal_load(&pl_row[plbase + 8 * (v >> 2) + 4 * g + (v & 3)]);
    hh[0] = __builtin_nontemporal_load(&ph_row[phbase + 2 * g]);
    hh[1] = __builtin_nontemporal_load(&ph_row[phbase + 2 * g + 1]);
    hh[2] = __builtin_nontemporal_load(&ph_row[phbase + 4 + 2 * g]);
    hh[3] = __builtin_nontemporal_load(&ph_row[phbase + 4 + 2 * g + 1]);
}

__device__ __forceinline__ v8i unpack_b(const v4i (&lo)[8], const v4i (&hh)[4]) {
    v8i b;
    b[0] = pack_q4(hh[0].x, hh[0].y, lo[0]);
    b[1] = pack_q4(hh[0].z, hh[0].w, lo[1]);
    b[2] = pack_q4(hh[1].x, hh[1].y, lo[2]);
    b[3] = pack_q4(hh[1].z, hh[1].w, lo[3]);
    b[4] = pack_q4(hh[2].x, hh[2].y, lo[4]);
    b[5] = pack_q4(hh[2].z, hh[2].w, lo[5]);
    b[6] = pack_q4(hh[3].x, hh[3].y, lo[6]);
    b[7] = pack_q4(hh[3].z, hh[3].w, lo[7]);
    return b;
}

__device__ __forceinline__ void wmma_tiles(const int* lds_base, int lane, v8i b, v8i (&acc)[16]) {
#pragma unroll
    for (int t = 0; t < 16; ++t) {
        v8i a = *(const v8i*)(lds_base + (t * 32 + lane) * 8);
        acc[t] = __builtin_amdgcn_wmma_i32_16x16x64_iu8(
             true, a,  true, b, acc[t],
             false,  false);
    }
}

__global__ __launch_bounds__(128)
void gemm_kernel(const int* __restrict__ a8, const int* __restrict__ ssum,
                 const int* __restrict__ packed_high, const int* __restrict__ packed_low,
                 const float* __restrict__ scale, const float* __restrict__ zp,
                 const float* __restrict__ bias, const float* __restrict__ act_scale,
                 float* __restrict__ out) {
    __shared__ __align__(32) int lds_ab[2][4096];
    __shared__ float lds_s[TOK];

    const int tid  = threadIdx.x;
    const int wave = tid >> 5;
    const int lane = tid & 31;
    const int g    = lane >> 4;
    const int o    = blockIdx.x * 64 + wave * 16 + (lane & 15);

    lds_s[tid]       = (float)ssum[tid];
    lds_s[tid + 128] = (float)ssum[tid + 128];

    v8i acc[16] = {};

    const v4i* __restrict__ ph_row = (const v4i*)(packed_high + (size_t)o * (KIN / 2));
    const v4i* __restrict__ pl_row = (const v4i*)(packed_low  + (size_t)o * KIN);

    v4i loA[8], hhA[4], loB[8], hhB[4];

    issue_slab(a8, 0, tid, &lds_ab[0][0]);
    load_w(ph_row, pl_row, 0, g, loA, hhA);

    for (int k0 = 0; k0 < KIN; k0 += 128) {
        const int s = k0 >> 6;

        issue_slab(a8, s + 1, tid, &lds_ab[1][0]);
        load_w(ph_row, pl_row, k0 + 64, g, loB, hhB);
        {
            v8i b = unpack_b(loA, hhA);
            wait_async_le8();
            __syncthreads();
            wmma_tiles(&lds_ab[0][0], lane, b, acc);
            __syncthreads();
        }

        const bool more = (k0 + 128) < KIN;
        if (more) {
            issue_slab(a8, s + 2, tid, &lds_ab[0][0]);
            load_w(ph_row, pl_row, k0 + 128, g, loA, hhA);
        }
        {
            v8i b = unpack_b(loB, hhB);
            if (more) wait_async_le8(); else wait_async_0();
            __syncthreads();
            wmma_tiles(&lds_ab[1][0], lane, b, acc);
            __syncthreads();
        }
    }

    const float cs = act_scale[0] * scale[o];
    const float zO = zp[o];
    const float bO = bias[o];
#pragma unroll
    for (int t = 0; t < 16; ++t) {
#pragma unroll
        for (int r = 0; r < 8; ++r) {
            const int tok = t * 16 + g * 8 + r;
            float v = cs * ((float)acc[t][r] - zO * lds_s[tok]) + bO;
            __builtin_nontemporal_store(v, &out[(size_t)tok * NOUT + o]);
        }
    }
}

extern "C" void kernel_launch(void* const* d_in, const int* in_sizes, int n_in,
                              void* d_out, int out_size, void* d_ws, size_t ws_size,
                              hipStream_t stream) {
    const float* x           = (const float*)d_in[0];
    const int*   packed_high = (const int*)d_in[1];
    const int*   packed_low  = (const int*)d_in[2];
    const float* scale       = (const float*)d_in[3];
    const float* zero_point  = (const float*)d_in[4];
    const float* bias        = (const float*)d_in[5];
    const float* act_scale   = (const float*)d_in[6];

    int* a8   = (int*)d_ws;
    int* ssum = (int*)((char*)d_ws + (size_t)TOK * KIN);

    quant_x_kernel<<<TOK, 256, 0, stream>>>(x, act_scale, a8, ssum);
    gemm_kernel<<<NOUT / 64, 128, 0, stream>>>(a8, ssum, packed_high, packed_low,
                                               scale, zero_point, bias, act_scale,
                                               (float*)d_out);
}
